// _WindowAttention1D_58815282151996
// MI455X (gfx1250) — matrix operand order repaired and checked by reading, not compiled, not run on hardware
//
#include <hip/hip_runtime.h>

typedef _Float16 v16h __attribute__((ext_vector_type(16)));
typedef _Float16 h8   __attribute__((ext_vector_type(8)));
typedef _Float16 h4   __attribute__((ext_vector_type(4)));
typedef float    v8f  __attribute__((ext_vector_type(8)));

#define DIMX 256
#define NHEAD 8
#define HDIM 32
#define WIN 64
#define NWIN 4096

union V16U { v16h v; h8 h[2]; };

static __device__ __forceinline__ v8f wmma16(v16h a, v16h b, v8f c) {
    return __builtin_amdgcn_wmma_f32_16x16x32_f16(
        false, a, false, b, (short)0, c, false, false);
}

// A fragment (16x32 f16) from row-major src. ISA layout: lanes 0-15 hold
// K=[0..7] in v0-3 and K=[16..23] in v4-7; lanes 16-31 hold K=[8..15],[24..31].
static __device__ __forceinline__ v16h loadA(const _Float16* base, int stride,
                                             int row0, int col0, int lr, int hi) {
    const _Float16* p = base + (row0 + lr) * stride + col0 + hi * 8;
    V16U u;
    u.h[0] = *(const h8*)(p);
    u.h[1] = *(const h8*)(p + 16);
    return u.v;
}

// B fragment (32x16 f16) from transposed (N-major) storage bt[n][k].
// Lane L: column n0 + L%16, K=[8h..8h+7] then [16+8h..16+8h+7], h=L/16.
static __device__ __forceinline__ v16h loadB(const _Float16* bt, int stride,
                                             int n0, int k0, int lr, int hi) {
    const _Float16* p = bt + (n0 + lr) * stride + k0 + hi * 8;
    V16U u;
    u.h[0] = *(const h8*)(p);
    u.h[1] = *(const h8*)(p + 16);
    return u.v;
}

// Prologue: f32 weights -> f16, transposed to N-major (B-fragment layout).
__global__ void prep_weights(const float* __restrict__ wqkv,
                             const float* __restrict__ wproj,
                             _Float16* __restrict__ wqkvt,
                             _Float16* __restrict__ wprojt) {
    int i = blockIdx.x * 256 + threadIdx.x;
    if (i < 3 * DIMX * DIMX) {
        int n = i >> 8, k = i & 255;
        wqkvt[i] = (_Float16)wqkv[k * (3 * DIMX) + n];
    }
    if (i < DIMX * DIMX) {
        int n = i >> 8, k = i & 255;
        wprojt[i] = (_Float16)wproj[k * DIMX + n];
    }
}

__global__ __launch_bounds__(256)
void win_attn_kernel(const float* __restrict__ x,
                     const float* __restrict__ b_qkv,
                     const float* __restrict__ b_proj,
                     const float* __restrict__ rel_bias,
                     const _Float16* __restrict__ wqkvt,
                     const _Float16* __restrict__ wprojt,
                     float* __restrict__ out) {
    // LDS layout (196KB): region0 = X(32KB, phase1) overlaid with P(64KB, ph2-3)
    __shared__ __align__(16) char smem[200704];
    _Float16* sX  = (_Float16*)smem;                    // 64x256 f16
    _Float16* sP  = (_Float16*)smem;                    // 8 x 64x64 f16
    _Float16* sQ  = (_Float16*)(smem + 65536);          // 8 x 64x32
    _Float16* sK  = (_Float16*)(smem + 98304);          // 8 x 64x32
    _Float16* sVt = (_Float16*)(smem + 131072);         // 8 x 32x64 (transposed)
    _Float16* sO  = (_Float16*)(smem + 163840);         // 64x256
    float*    sB  = (float*)(smem + 196608);            // 8 x 128 bias

    const int tid  = threadIdx.x;
    const int lane = tid & 31;
    const int wv   = tid >> 5;       // wave id == head id
    const int lr   = lane & 15;
    const int hi   = lane >> 4;
    const int win  = blockIdx.x;

    // ---- Phase 0: X window -> LDS (f16), bias -> LDS ----
    {
        const float4* xg = (const float4*)(x + (size_t)win * WIN * DIMX);
        for (int i = tid; i < WIN * DIMX / 4; i += 256) {
            float4 f = xg[i];
            h4 hh = { (_Float16)f.x, (_Float16)f.y, (_Float16)f.z, (_Float16)f.w };
            ((h4*)sX)[i] = hh;
        }
        for (int i = tid; i < NHEAD * 127; i += 256)
            sB[(i / 127) * 128 + (i % 127)] = rel_bias[i];
    }
    __syncthreads();

    const int h = wv;
    const float scale = 0.17677669529663689f;  // 1/sqrt(32)
    _Float16* q_h  = sQ  + h * (WIN * HDIM);
    _Float16* k_h  = sK  + h * (WIN * HDIM);
    _Float16* vt_h = sVt + h * (HDIM * WIN);
    _Float16* p_h  = sP  + h * (WIN * WIN);
    const float* bh = sB + h * 128;

    float bq0 = b_qkv[h * HDIM + lr],            bq1 = b_qkv[h * HDIM + 16 + lr];
    float bk0 = b_qkv[DIMX + h * HDIM + lr],     bk1 = b_qkv[DIMX + h * HDIM + 16 + lr];
    float bv0 = b_qkv[2 * DIMX + h * HDIM + lr], bv1 = b_qkv[2 * DIMX + h * HDIM + 16 + lr];

    // ---- Phase 1: q/k/v = X @ Wqkv (per head) ----
    for (int mi = 0; mi < 4; ++mi) {
        v8f cq0 = {}, cq1 = {}, ck0 = {}, ck1 = {}, cv0 = {}, cv1 = {};
        for (int kk = 0; kk < 8; ++kk) {
            v16h a = loadA(sX, DIMX, mi * 16, kk * 32, lr, hi);
            cq0 = wmma16(a, loadB(wqkvt, DIMX, h * HDIM,                 kk * 32, lr, hi), cq0);
            cq1 = wmma16(a, loadB(wqkvt, DIMX, h * HDIM + 16,            kk * 32, lr, hi), cq1);
            ck0 = wmma16(a, loadB(wqkvt, DIMX, DIMX + h * HDIM,          kk * 32, lr, hi), ck0);
            ck1 = wmma16(a, loadB(wqkvt, DIMX, DIMX + h * HDIM + 16,     kk * 32, lr, hi), ck1);
            cv0 = wmma16(a, loadB(wqkvt, DIMX, 2 * DIMX + h * HDIM,      kk * 32, lr, hi), cv0);
            cv1 = wmma16(a, loadB(wqkvt, DIMX, 2 * DIMX + h * HDIM + 16, kk * 32, lr, hi), cv1);
        }
#pragma unroll
        for (int r = 0; r < 8; ++r) {
            int m = mi * 16 + r + 8 * hi;
            q_h[m * HDIM + lr]        = (_Float16)((cq0[r] + bq0) * scale);
            q_h[m * HDIM + 16 + lr]   = (_Float16)((cq1[r] + bq1) * scale);
            k_h[m * HDIM + lr]        = (_Float16)(ck0[r] + bk0);
            k_h[m * HDIM + 16 + lr]   = (_Float16)(ck1[r] + bk1);
            vt_h[lr * WIN + m]        = (_Float16)(cv0[r] + bv0);
            vt_h[(16 + lr) * WIN + m] = (_Float16)(cv1[r] + bv1);
        }
    }
    __syncthreads();  // all waves done reading sX; sP region reusable

    // ---- Phase 2: S = q k^T (+bias), softmax in-register, P -> LDS ----
    for (int mi = 0; mi < 4; ++mi) {
        v8f s0 = {}, s1 = {}, s2 = {}, s3 = {};
        v16h aq = loadA(q_h, HDIM, mi * 16, 0, lr, hi);
        s0 = wmma16(aq, loadB(k_h, HDIM, 0,  0, lr, hi), s0);
        s1 = wmma16(aq, loadB(k_h, HDIM, 16, 0, lr, hi), s1);
        s2 = wmma16(aq, loadB(k_h, HDIM, 32, 0, lr, hi), s2);
        s3 = wmma16(aq, loadB(k_h, HDIM, 48, 0, lr, hi), s3);
#pragma unroll
        for (int r = 0; r < 8; ++r) {
            int row = mi * 16 + r + 8 * hi;
            float v0 = s0[r] + bh[lr      - row + 63];
            float v1 = s1[r] + bh[lr + 16 - row + 63];
            float v2 = s2[r] + bh[lr + 32 - row + 63];
            float v3 = s3[r] + bh[lr + 48 - row + 63];
            float mx = fmaxf(fmaxf(v0, v1), fmaxf(v2, v3));
            mx = fmaxf(mx, __shfl_xor(mx, 1, 32));
            mx = fmaxf(mx, __shfl_xor(mx, 2, 32));
            mx = fmaxf(mx, __shfl_xor(mx, 4, 32));
            mx = fmaxf(mx, __shfl_xor(mx, 8, 32));
            v0 = __expf(v0 - mx); v1 = __expf(v1 - mx);
            v2 = __expf(v2 - mx); v3 = __expf(v3 - mx);
            float sm = v0 + v1 + v2 + v3;
            sm += __shfl_xor(sm, 1, 32);
            sm += __shfl_xor(sm, 2, 32);
            sm += __shfl_xor(sm, 4, 32);
            sm += __shfl_xor(sm, 8, 32);
            float rn = 1.0f / sm;
            p_h[row * WIN + lr]      = (_Float16)(v0 * rn);
            p_h[row * WIN + lr + 16] = (_Float16)(v1 * rn);
            p_h[row * WIN + lr + 32] = (_Float16)(v2 * rn);
            p_h[row * WIN + lr + 48] = (_Float16)(v3 * rn);
        }
    }

    // ---- Phase 3: O = P @ V -> shared O (f16) ----
    for (int mi = 0; mi < 4; ++mi) {
        v8f o0 = {}, o1 = {};
        for (int kk = 0; kk < 2; ++kk) {
            v16h ap = loadA(p_h, WIN, mi * 16, kk * 32, lr, hi);
            o0 = wmma16(ap, loadB(vt_h, WIN, 0,  kk * 32, lr, hi), o0);
            o1 = wmma16(ap, loadB(vt_h, WIN, 16, kk * 32, lr, hi), o1);
        }
#pragma unroll
        for (int r = 0; r < 8; ++r) {
            int m = mi * 16 + r + 8 * hi;
            sO[m * DIMX + h * HDIM + lr]      = (_Float16)o0[r];
            sO[m * DIMX + h * HDIM + 16 + lr] = (_Float16)o1[r];
        }
    }
    __syncthreads();

    // ---- Phase 4: out = O @ Wproj + b_proj (waves split output columns) ----
    const int c0 = wv * 32;
    float bp0 = b_proj[c0 + lr];
    float bp1 = b_proj[c0 + 16 + lr];
    for (int mi = 0; mi < 4; ++mi) {
        v8f d0 = {}, d1 = {};
        for (int kk = 0; kk < 8; ++kk) {
            v16h a = loadA(sO, DIMX, mi * 16, kk * 32, lr, hi);
            d0 = wmma16(a, loadB(wprojt, DIMX, c0,      kk * 32, lr, hi), d0);
            d1 = wmma16(a, loadB(wprojt, DIMX, c0 + 16, kk * 32, lr, hi), d1);
        }
        float* og = out + ((size_t)win * WIN + mi * 16) * DIMX;
#pragma unroll
        for (int r = 0; r < 8; ++r) {
            int m = r + 8 * hi;
            og[m * DIMX + c0 + lr]      = d0[r] + bp0;
            og[m * DIMX + c0 + 16 + lr] = d1[r] + bp1;
        }
    }
}

extern "C" void kernel_launch(void* const* d_in, const int* in_sizes, int n_in,
                              void* d_out, int out_size, void* d_ws, size_t ws_size,
                              hipStream_t stream) {
    const float* x      = (const float*)d_in[0];
    const float* w_qkv  = (const float*)d_in[1];
    const float* b_qkv  = (const float*)d_in[2];
    const float* w_proj = (const float*)d_in[3];
    const float* b_proj = (const float*)d_in[4];
    const float* rel    = (const float*)d_in[5];

    _Float16* wqkvt  = (_Float16*)d_ws;                    // 768x256 f16 (N-major)
    _Float16* wprojt = wqkvt + 3 * DIMX * DIMX;            // 256x256 f16 (N-major)

    prep_weights<<<(3 * DIMX * DIMX + 255) / 256, 256, 0, stream>>>(
        w_qkv, w_proj, wqkvt, wprojt);
    win_attn_kernel<<<NWIN, 256, 0, stream>>>(
        x, b_qkv, b_proj, rel, wqkvt, wprojt, (float*)d_out);
}
